// SSM_55198919688744
// MI455X (gfx1250) — compile-verified, matrix operand order checked by reading, not run on hardware
//
#include <hip/hip_runtime.h>

typedef __attribute__((ext_vector_type(16))) __bf16 v16bf;
typedef __attribute__((ext_vector_type(8)))  float  v8f;

#define TTOK 4096      // B*S tokens
#define DMODEL 256
#define DSTATE 64
#define NA 4096        // DS*DS
#define NB 16384       // DS*DM

// ---------- bf16 helpers (bit-level, no reliance on __bf16 conversions) ----------
static __device__ inline unsigned short f2bf_bits(float f) {
  union { float f; unsigned u; } v; v.f = f;
  unsigned r = v.u + 0x7fffu + ((v.u >> 16) & 1u);   // round-to-nearest-even
  return (unsigned short)(r >> 16);
}
static __device__ inline float bf2f(unsigned short b) {
  union { unsigned u; float f; } v; v.u = ((unsigned)b) << 16;
  return v.f;
}
// K-offset inside a 16x32 bf16 WMMA operand fragment for (half j, lane L)
static __device__ inline int fklane(int j, int lane) {
  return ((j >> 3) << 4) + ((lane >> 4) << 3) + (j & 7);
}

// ---------- CDNA5 async global -> LDS copy (ASYNCcnt-tracked DMA path) ----------
// LDS address from a VGPR; global address = SGPR base + 32-bit VGPR byte offset
static __device__ inline void async_b128(unsigned lds_off, const void* base, unsigned byte_off) {
  asm volatile("global_load_async_to_lds_b128 %0, %1, %2"
               :: "v"(lds_off), "v"(byte_off),
                  "s"((unsigned long long)(size_t)base)
               : "memory");
}
static __device__ inline void async_wait0() {
  asm volatile("s_wait_asynccnt 0" ::: "memory");
}

// ---------- pack x [T,256] f32 -> bf16 A-operand fragments ----------
// layout: [mtile(256)][ktile(8)][lane(32)][half(16)]
__global__ void pack_x_k(const float* __restrict__ x, unsigned short* __restrict__ xp) {
  int idx  = blockIdx.x * blockDim.x + threadIdx.x;
  int elem = idx & 511, frag = idx >> 9;
  int lane = elem >> 4, j = elem & 15;
  int kt = frag & 7, mtile = frag >> 3;
  int m = mtile * 16 + (lane & 15);
  int k = kt * 32 + fklane(j, lane);
  xp[idx] = f2bf_bits(x[m * DMODEL + k]);
}

// ---------- pack W [256,N] f32 (row-major) -> bf16 B-operand fragments ----------
// layout: [ntile(N/16)][ktile(8)][lane(32)][half(16)]
__global__ void pack_w_k(const float* __restrict__ wsrc, unsigned short* __restrict__ wp, int N) {
  int idx  = blockIdx.x * blockDim.x + threadIdx.x;
  int elem = idx & 511, frag = idx >> 9;
  int lane = elem >> 4, j = elem & 15;
  int kt = frag & 7, ntile = frag >> 3;
  int n = ntile * 16 + (lane & 15);
  int k = kt * 32 + fklane(j, lane);
  wp[idx] = f2bf_bits(wsrc[k * N + n]);
}

// ---------- pack WC [256, 64*256] into (n,k)-flattened GEMM operand ----------
// WC2[(n*256+k), d] = WC[k, n*256+d]; layout [ntile(16)][ktile(512)][lane][half]
__global__ void pack_wc2_k(const float* __restrict__ wc, unsigned short* __restrict__ wp) {
  int idx  = blockIdx.x * blockDim.x + threadIdx.x;
  int elem = idx & 511, frag = idx >> 9;
  int lane = elem >> 4, j = elem & 15;
  int kt = frag & 511, ntile = frag >> 9;
  int d  = ntile * 16 + (lane & 15);
  int kk = kt * 32 + fklane(j, lane);
  int n = kk >> 8, k = kk & 255;
  wp[idx] = f2bf_bits(wc[k * NB + n * DMODEL + d]);
}

// ---------- logD = x @ WD + bD  [T,64] (tiny GEMV, f32 VALU) ----------
__global__ void logd_k(const float* __restrict__ x, const float* __restrict__ WD,
                       const float* __restrict__ bD, float* __restrict__ logD) {
  int t = blockIdx.x, n = threadIdx.x;
  float s = bD[n];
  for (int k = 0; k < DMODEL; ++k) s += x[t * DMODEL + k] * WD[k * DSTATE + n];
  logD[t * DSTATE + n] = s;
}

// ---------- A = (x@WA + bA) * exp(logD) ; WMMA GEMM 4096x4096x256 ----------
// 4 waves share one 64-col B panel staged once into LDS via async copies.
__global__ void __launch_bounds__(128) gemmA_k(const unsigned short* __restrict__ xp,
                                               const unsigned short* __restrict__ wap,
                                               const float* __restrict__ bA,
                                               const float* __restrict__ logD,
                                               float* __restrict__ A) {
  __shared__ __align__(16) unsigned short bsh[16384];   // 4 ntiles x 8 kt x 512 = 32KB
  int tid = threadIdx.x;
  int w = tid >> 5, lane = tid & 31;
  int mtile = blockIdx.y * 4 + w;        // per-wave token tile
  int ntb   = blockIdx.x * 4;            // shared 64-col panel
  unsigned lb = (unsigned)(size_t)(&bsh[0]);
#pragma unroll
  for (int r = 0; r < 16; ++r) {         // 2048 x 16B units
    int u = r * 128 + tid;
    int f = u >> 9, within = u & 511;
    async_b128(lb + (unsigned)(u * 16), wap,
               (unsigned)(((ntb + f) * 4096 + within * 8) * 2));
  }
  async_wait0();
  __syncthreads();

  v8f acc[4] = {};
  for (int kt = 0; kt < 8; ++kt) {
    v16bf a = *(const v16bf*)(xp + (((mtile * 8 + kt) << 9) + lane * 16));
#pragma unroll
    for (int f = 0; f < 4; ++f) {
      v16bf b = *(const v16bf*)(&bsh[f * 4096 + kt * 512 + lane * 16]);
      acc[f] = __builtin_amdgcn_wmma_f32_16x16x32_bf16(false, a, false, b,
                                                       (short)0, acc[f], false, false);
    }
  }
  int hi = lane >> 4, ln = lane & 15;
#pragma unroll
  for (int f = 0; f < 4; ++f) {
    int col = (ntb + f) * 16 + ln;
    float bias = bA[col];
    int nouter = col >> 6;
#pragma unroll
    for (int r = 0; r < 8; ++r) {
      int t = mtile * 16 + r + hi * 8;
      float sc = __expf(logD[t * DSTATE + nouter]);
      A[(size_t)t * NA + col] = (acc[f][r] + bias) * sc;
    }
  }
}

// ---------- Bx[t,n] = sum_d (x@WB + bB)[t, n*256+d] * x[t,d]  (fused) ----------
// 4 waves (4 token tiles) share one n; WB fragments double-buffered through LDS.
__global__ void __launch_bounds__(128) bx_k(const unsigned short* __restrict__ xp,
                                            const unsigned short* __restrict__ wbp,
                                            const float* __restrict__ x,
                                            const float* __restrict__ bB,
                                            float* __restrict__ Bx) {
  __shared__ __align__(16) unsigned short bsh[2][4096];  // 2 x 8KB
  int tid = threadIdx.x;
  int w = tid >> 5, lane = tid & 31;
  int n   = blockIdx.x & 63;             // shared state row
  int grp = blockIdx.x >> 6;
  int mtile = grp * 4 + w;               // per-wave token tile
  unsigned lb = (unsigned)(size_t)(&bsh[0][0]);

  v16bf a[8];
#pragma unroll
  for (int kt = 0; kt < 8; ++kt)
    a[kt] = *(const v16bf*)(xp + (((mtile * 8 + kt) << 9) + lane * 16));
  int hi = lane >> 4, ln = lane & 15;

  // prologue: stage dsub=0 into buffer 0
#pragma unroll
  for (int r = 0; r < 4; ++r) {
    int u = r * 128 + tid;               // 512 x 16B units = 8KB
    async_b128(lb + (unsigned)(u * 16), wbp,
               (unsigned)(((n * 16 + 0) * 4096 + u * 8) * 2));
  }

  float tot[8] = {};
  for (int dsub = 0; dsub < 16; ++dsub) {
    int buf = dsub & 1;
    async_wait0();
    __syncthreads();                     // staged data visible; prev compute done
    if (dsub + 1 < 16) {
      int nb = (dsub + 1) & 1;
#pragma unroll
      for (int r = 0; r < 4; ++r) {
        int u = r * 128 + tid;
        async_b128(lb + (unsigned)(nb * 8192 + u * 16), wbp,
                   (unsigned)(((n * 16 + dsub + 1) * 4096 + u * 8) * 2));
      }
    }
    v8f acc = {};
#pragma unroll
    for (int kt = 0; kt < 8; ++kt) {
      v16bf b = *(const v16bf*)(&bsh[buf][kt * 512 + lane * 16]);
      acc = __builtin_amdgcn_wmma_f32_16x16x32_bf16(false, a[kt], false, b,
                                                    (short)0, acc, false, false);
    }
    int dd = dsub * 16 + ln;
    float bb = bB[n * DMODEL + dd];
#pragma unroll
    for (int r = 0; r < 8; ++r) {
      int t = mtile * 16 + r + hi * 8;
      tot[r] += (acc[r] + bb) * x[t * DMODEL + dd];
    }
    __syncthreads();                     // all waves done reading buf before overwrite
  }
#pragma unroll
  for (int r = 0; r < 8; ++r) {          // reduce 16 column-lanes per half
    float v = tot[r];
    v += __shfl_xor(v, 1, 32);
    v += __shfl_xor(v, 2, 32);
    v += __shfl_xor(v, 4, 32);
    v += __shfl_xor(v, 8, 32);
    if (ln == 0) {
      int t = mtile * 16 + r + hi * 8;
      Bx[t * DSTATE + n] = v;
    }
  }
}

// ---------- sequential scan: h = A_t @ h + Bx_t (per batch), h in LDS ----------
__global__ void __launch_bounds__(256) scan_k(const float* __restrict__ A,
                                              const float* __restrict__ Bx,
                                              float* __restrict__ hseq) {
  __shared__ float h[DSTATE];
  __shared__ float part[256];
  int b = blockIdx.x, tid = threadIdx.x;
  int i = tid >> 2, g = tid & 3;
  if (tid < DSTATE) h[tid] = 0.f;
  __syncthreads();
  for (int s = 0; s < 1024; ++s) {
    int t = b * 1024 + s;
    const float4* Arow = (const float4*)(A + (size_t)t * NA + i * 64 + g * 16);
    float ps = 0.f;
#pragma unroll
    for (int q = 0; q < 4; ++q) {
      float4 av = Arow[q];
      const float* hp = h + g * 16 + q * 4;
      ps += av.x * hp[0] + av.y * hp[1] + av.z * hp[2] + av.w * hp[3];
    }
    part[tid] = ps;
    __syncthreads();
    float hn = 0.f;
    if (tid < DSTATE)
      hn = part[tid * 4] + part[tid * 4 + 1] + part[tid * 4 + 2] + part[tid * 4 + 3]
         + Bx[t * DSTATE + tid];
    __syncthreads();
    if (tid < DSTATE) { h[tid] = hn; hseq[t * DSTATE + tid] = hn; }
    __syncthreads();
  }
}

// ---------- out[t,d] = sum_{n,k} h[t,n] x[t,k] WC[k,n*256+d] + bC-term ----------
// A-operand (h outer x) synthesized in registers; WC panels double-buffered via async LDS.
__global__ void __launch_bounds__(128) out_k(const unsigned short* __restrict__ xp,
                                             const unsigned short* __restrict__ wcp,
                                             const float* __restrict__ hseq,
                                             const float* __restrict__ bC,
                                             float* __restrict__ out) {
  __shared__ __align__(16) unsigned short bsh[2][16384]; // 2 x 32KB
  int tid = threadIdx.x;
  int w = tid >> 5, lane = tid & 31;
  int ng  = blockIdx.x >> 6;             // shared 64-col output panel
  int grp = blockIdx.x & 63;
  int mtile = grp * 4 + w;               // per-wave token tile
  int hi = lane >> 4, ln = lane & 15;
  unsigned lb = (unsigned)(size_t)(&bsh[0][0]);

  v16bf xf[8];
#pragma unroll
  for (int kx = 0; kx < 8; ++kx)
    xf[kx] = *(const v16bf*)(xp + (((mtile * 8 + kx) << 9) + lane * 16));

  // prologue: stage n=0 panel (4 ntiles x 8 kt = 32KB) into buffer 0
#pragma unroll
  for (int r = 0; r < 16; ++r) {
    int u = r * 128 + tid;               // 2048 x 16B units
    int f = u >> 9, within = u & 511;
    async_b128(lb + (unsigned)(u * 16), wcp,
               (unsigned)((((ng * 4 + f) * 512 + 0 * 8) * 512 + within * 8) * 2));
  }

  v8f acc[4] = {};
  int trow = mtile * 16 + ln;            // A-operand row for this lane
  for (int n = 0; n < DSTATE; ++n) {
    int buf = n & 1;
    async_wait0();
    __syncthreads();
    if (n + 1 < DSTATE) {
      int nb = (n + 1) & 1;
#pragma unroll
      for (int r = 0; r < 16; ++r) {
        int u = r * 128 + tid;
        int f = u >> 9, within = u & 511;
        async_b128(lb + (unsigned)(nb * 32768 + u * 16), wcp,
                   (unsigned)((((ng * 4 + f) * 512 + (n + 1) * 8) * 512 + within * 8) * 2));
      }
    }
    float hv = hseq[trow * DSTATE + n];
    for (int kx = 0; kx < 8; ++kx) {
      union { v16bf v; unsigned short s[16]; } af, xin;
      xin.v = xf[kx];
#pragma unroll
      for (int j = 0; j < 16; ++j)
        af.s[j] = f2bf_bits(hv * bf2f(xin.s[j]));
#pragma unroll
      for (int f = 0; f < 4; ++f) {
        v16bf b = *(const v16bf*)(&bsh[buf][f * 4096 + kx * 512 + lane * 16]);
        acc[f] = __builtin_amdgcn_wmma_f32_16x16x32_bf16(false, af.v, false, b,
                                                         (short)0, acc[f], false, false);
      }
    }
    __syncthreads();                     // all waves done with buf before overwrite
  }
#pragma unroll
  for (int r = 0; r < 8; ++r) {
    int t = mtile * 16 + r + hi * 8;
    float v[4];
#pragma unroll
    for (int f = 0; f < 4; ++f) v[f] = acc[f][r];
    for (int n = 0; n < DSTATE; ++n) {
      float hv2 = hseq[t * DSTATE + n];
#pragma unroll
      for (int f = 0; f < 4; ++f) {
        int d = (ng * 4 + f) * 16 + ln;
        v[f] += bC[n * DMODEL + d] * hv2;
      }
    }
#pragma unroll
    for (int f = 0; f < 4; ++f) {
      int d = (ng * 4 + f) * 16 + ln;
      out[t * DMODEL + d] = v[f];
    }
  }
}

extern "C" void kernel_launch(void* const* d_in, const int* in_sizes, int n_in,
                              void* d_out, int out_size, void* d_ws, size_t ws_size,
                              hipStream_t stream) {
  const float* x  = (const float*)d_in[0];
  const float* WA = (const float*)d_in[1];
  const float* bA = (const float*)d_in[2];
  const float* WB = (const float*)d_in[3];
  const float* bB = (const float*)d_in[4];
  const float* WC = (const float*)d_in[5];
  const float* bC = (const float*)d_in[6];
  const float* WD = (const float*)d_in[7];
  const float* bD = (const float*)d_in[8];
  // d_in[9] = Wdelta, d_in[10] = bdelta: delta is dead code in the reference.
  float* out = (float*)d_out;

  char* wsb = (char*)d_ws;
  unsigned short* xp   = (unsigned short*)(wsb + (size_t)0);            //  2 MB
  unsigned short* wap  = (unsigned short*)(wsb + ((size_t)2  << 20));   //  2 MB
  unsigned short* wbp  = (unsigned short*)(wsb + ((size_t)4  << 20));   //  8 MB
  unsigned short* wcp  = (unsigned short*)(wsb + ((size_t)12 << 20));   //  8 MB
  float*          logD = (float*)(wsb + ((size_t)20 << 20));            //  1 MB
  float*          Bx   = (float*)(wsb + ((size_t)21 << 20));            //  1 MB
  float*          hseq = (float*)(wsb + ((size_t)22 << 20));            //  1 MB
  float*          A    = (float*)(wsb + ((size_t)23 << 20));            // 64 MB

  pack_x_k  <<<4096,  256, 0, stream>>>(x,  xp);
  pack_w_k  <<<4096,  256, 0, stream>>>(WA, wap, NA);
  pack_w_k  <<<16384, 256, 0, stream>>>(WB, wbp, NB);
  pack_wc2_k<<<16384, 256, 0, stream>>>(WC, wcp);
  logd_k    <<<4096,  64,  0, stream>>>(x, WD, bD, logD);
  gemmA_k   <<<dim3(64, 64), 128, 0, stream>>>(xp, wap, bA, logD, A);
  bx_k      <<<4096,  128, 0, stream>>>(xp, wbp, x, bB, Bx);
  scan_k    <<<4,     256, 0, stream>>>(A, Bx, hseq);
  out_k     <<<256,   128, 0, stream>>>(xp, wcp, hseq, bC, out);
}
